// HypercomplexSelfAttention_19799799235029
// MI455X (gfx1250) — matrix operand order repaired and checked by reading, not compiled, not run on hardware
//
#include <hip/hip_runtime.h>
#include <hip/hip_bf16.h>

typedef __bf16 bf16_t;
typedef __attribute__((ext_vector_type(16))) __bf16 v16bf;
typedef __attribute__((ext_vector_type(8)))  __bf16 v8bf;
typedef __attribute__((ext_vector_type(8)))  float  v8f;

#define DEM 256
#define SEQ 1024
#define NT  64                 // n-tile: per-tile working set ~170MB < 192MB L2
#define JK  (DEM*DEM)          // 65536
#define CTOT (SEQ*DEM)         // 262144

// ---------------- WMMA fragment helpers (bf16, 16x16x32) ----------------
// A-frag: matrix stored [M][K] row-major, contraction K contiguous.
// lane r=lane&15 -> row; elems t<8 : K=kb+t+8h ; t>=8 : K=kb+16+(t-8)+8h
__device__ __forceinline__ v16bf load_a_frag(const bf16_t* base, int row0,
                                             int stride, int kb, int lane) {
  int r = lane & 15, h = lane >> 4;
  const bf16_t* p = base + (size_t)(row0 + r) * stride + kb + 8*h;
  v8bf lo = *(const v8bf*)p;
  v8bf hi = *(const v8bf*)(p + 16);
  v16bf a;
#pragma unroll
  for (int t = 0; t < 8; ++t) { a[t] = lo[t]; a[t+8] = hi[t]; }
  return a;
}

// B-frag: mathematical B (KxN) stored transposed [N][K], K contiguous.
// lane r -> col N; elems t<8 : K=kb+t+8h ; t>=8 : K=kb+16+(t-8)+8h
__device__ __forceinline__ v16bf load_b_frag(const bf16_t* base, int col0,
                                             int stride, int kb, int lane) {
  int r = lane & 15, h = lane >> 4;
  const bf16_t* p = base + (size_t)(col0 + r) * stride + kb + 8*h;
  v8bf lo = *(const v8bf*)p;
  v8bf hi = *(const v8bf*)(p + 16);
  v16bf b;
#pragma unroll
  for (int t = 0; t < 8; ++t) { b[t] = lo[t]; b[t+8] = hi[t]; }
  return b;
}

__device__ __forceinline__ v8f wmma_bf16(v16bf a, v16bf b, v8f c) {
  return __builtin_amdgcn_wmma_f32_16x16x32_bf16(
      /*neg_a=*/false, a, /*neg_b=*/false, b,
      /*c_mod=*/(short)0, c, /*reuse_a=*/false, /*reuse_b=*/false);
}

// ---------------- small utility kernels ----------------
__global__ void __launch_bounds__(256) zero_f32(float* p, int n) {
  int i = blockIdx.x * 256 + threadIdx.x;
  if (i < n) p[i] = 0.0f;
}

// Q = X Wq^T + bq ; K = X Wk^T + bk ; store bf16 row-major [n][i]
__global__ void __launch_bounds__(256) qk_kernel(const float* __restrict__ X,
    const float* __restrict__ Wqw, const float* __restrict__ Wqb,
    const float* __restrict__ Wkw, const float* __restrict__ Wkb,
    bf16_t* __restrict__ Qb, bf16_t* __restrict__ Kb) {
  __shared__ float xs[DEM];
  int n = blockIdx.x, o = threadIdx.x;
  xs[o] = X[(size_t)n * DEM + o];
  __syncthreads();
  float q = Wqb[o], k = Wkb[o];
  const float4* wq = (const float4*)(Wqw + (size_t)o * DEM);
  const float4* wk = (const float4*)(Wkw + (size_t)o * DEM);
  const float4* xv = (const float4*)xs;
#pragma unroll 4
  for (int i = 0; i < DEM / 4; ++i) {
    float4 x4 = xv[i], a4 = wq[i], b4 = wk[i];
    q += x4.x*a4.x + x4.y*a4.y + x4.z*a4.z + x4.w*a4.w;
    k += x4.x*b4.x + x4.y*b4.y + x4.z*b4.z + x4.w*b4.w;
  }
  Qb[(size_t)n * DEM + o] = (bf16_t)q;
  Kb[(size_t)n * DEM + o] = (bf16_t)k;
}

// T[i][j][k] (f32) -> Tb2[(k*256+j)][i] (bf16): transpose + (j,k) digit swap
__global__ void __launch_bounds__(256) transpose_T(const float* __restrict__ T,
                                                   bf16_t* __restrict__ Tb2) {
  __shared__ float tile[32][33];
  int jk0 = blockIdx.x * 32, i0 = blockIdx.y * 32;
  int tx = threadIdx.x, ty = threadIdx.y;
#pragma unroll
  for (int s = 0; s < 4; ++s)
    tile[ty + 8*s][tx] = T[(size_t)(i0 + ty + 8*s) * JK + jk0 + tx];
  __syncthreads();
#pragma unroll
  for (int s = 0; s < 4; ++s) {
    int jk = jk0 + ty + 8*s;
    int j = jk >> 8, k = jk & 255;
    Tb2[(size_t)(k * DEM + j) * DEM + i0 + tx] = (bf16_t)tile[tx][ty + 8*s];
  }
}

// V[(m*256+k)][d] (f32) -> Vbt[d][(m*256+k)] (bf16)
__global__ void __launch_bounds__(256) transpose_V(const float* __restrict__ V,
                                                   bf16_t* __restrict__ Vbt) {
  __shared__ float tile[32][33];
  int c0 = blockIdx.x * 32, d0 = blockIdx.y * 32;
  int tx = threadIdx.x, ty = threadIdx.y;
#pragma unroll
  for (int s = 0; s < 4; ++s)
    tile[ty + 8*s][tx] = V[(size_t)(c0 + ty + 8*s) * DEM + d0 + tx];
  __syncthreads();
#pragma unroll
  for (int s = 0; s < 4; ++s)
    Vbt[(size_t)(d0 + ty + 8*s) * CTOT + c0 + tx] = (bf16_t)tile[tx][ty + 8*s];
}

// ---------------- phase 2: tmp_t[nl][k][j] = sum_i Q[n,i] T[i,j,k] ----------------
__global__ void __launch_bounds__(256) phase2_kernel(const bf16_t* __restrict__ Qb,
    const bf16_t* __restrict__ Tb2, bf16_t* __restrict__ tmpt, int n0) {
  int lane = threadIdx.x & 31, w = threadIdx.x >> 5;
  int nrow0 = n0 + blockIdx.x * 16;           // global Q rows
  int c2_0  = blockIdx.y * 128 + w * 16;      // c2 = k*256+j (j fastest)
  v8f c = {};
#pragma unroll
  for (int s = 0; s < 8; ++s) {
    v16bf a = load_a_frag(Qb,  nrow0, DEM, 32*s, lane);
    v16bf b = load_b_frag(Tb2, c2_0,  DEM, 32*s, lane);
    c = wmma_bf16(a, b, c);
  }
  int r = lane & 15, h = lane >> 4, nl0 = blockIdx.x * 16;
#pragma unroll
  for (int v = 0; v < 8; ++v)
    tmpt[(size_t)(nl0 + v + 8*h) * JK + c2_0 + r] = (bf16_t)c[v];
}

// ---------------- phase 3a: E[nl][m][k] = exp(A/16); Z[nl][k] = sum_m E ----------------
__global__ void __launch_bounds__(256) phase3a_kernel(const bf16_t* __restrict__ Kb,
    const bf16_t* __restrict__ tmpt, bf16_t* __restrict__ E, float* __restrict__ Z) {
  __shared__ float zbuf[64];
  int lane = threadIdx.x & 31, w = threadIdx.x >> 5;
  int m0 = blockIdx.x * 64, k0 = blockIdx.y * 64, nl = blockIdx.z;
  if (threadIdx.x < 64) zbuf[threadIdx.x] = 0.0f;
  __syncthreads();
  int mw  = m0 + (w & 3) * 16;
  int kg0 = k0 + (w >> 2) * 32;               // this wave: cols kg0..kg0+31
  const bf16_t* tbase = tmpt + (size_t)nl * JK;
  v8f c0 = {}, c1 = {};
#pragma unroll
  for (int s = 0; s < 8; ++s) {
    v16bf a  = load_a_frag(Kb, mw, DEM, 32*s, lane);
    v16bf b0 = load_b_frag(tbase, kg0,      DEM, 32*s, lane);
    c0 = wmma_bf16(a, b0, c0);
    v16bf b1 = load_b_frag(tbase, kg0 + 16, DEM, 32*s, lane);
    c1 = wmma_bf16(a, b1, c1);
  }
  int r = lane & 15, h = lane >> 4;
  bf16_t* ebase = E + (size_t)nl * CTOT;
  float z0 = 0.0f, z1 = 0.0f;
#pragma unroll
  for (int v = 0; v < 8; ++v) {
    float e0 = __expf(c0[v] * 0.0625f);       // 1/sqrt(256)
    float e1 = __expf(c1[v] * 0.0625f);
    int m = mw + v + 8*h;
    ebase[(size_t)m * DEM + kg0 + r]      = (bf16_t)e0;
    ebase[(size_t)m * DEM + kg0 + 16 + r] = (bf16_t)e1;
    z0 += e0; z1 += e1;
  }
  atomicAdd(&zbuf[kg0 - k0 + r],      z0);
  atomicAdd(&zbuf[kg0 - k0 + 16 + r], z1);
  __syncthreads();
  if (threadIdx.x < 64)
    atomicAdd(&Z[nl * DEM + k0 + threadIdx.x], zbuf[threadIdx.x]);
}

// E /= Z broadcast over m (in place, L2-hot)
__global__ void __launch_bounds__(256) normalize_kernel(bf16_t* __restrict__ E,
                                                        const float* __restrict__ Z) {
  size_t base = ((size_t)blockIdx.x * 256 + threadIdx.x) * 8;
  int nl = (int)(base >> 18);                 // / CTOT
  int cc = (int)(base & (CTOT - 1));
  int k  = cc & 255;                          // 8 consecutive k, same m row
  v8bf e = *(v8bf*)(E + base);
  const float* zp = Z + nl * DEM + k;
  v8bf o;
#pragma unroll
  for (int t = 0; t < 8; ++t)
    o[t] = (bf16_t)((float)e[t] / zp[t]);
  *(v8bf*)(E + base) = o;
}

// ---------------- phase 3b: out[n][d] += attn[n][(m,k)] @ Vbt[d][(m,k)] ----------------
// Each wave computes TWO 16x16 d-tiles sharing one A-fragment: two independent
// WMMA accumulation chains for the XDL pipe, half the A-side load traffic.
__global__ void __launch_bounds__(256) phase3b_kernel(const bf16_t* __restrict__ E,
    const bf16_t* __restrict__ Vbt, float* __restrict__ out, int n0) {
  int lane = threadIdx.x & 31, w = threadIdx.x >> 5;
  int nl0 = blockIdx.x * 16;
  int dA  = w * 16;                            // cols 0..127
  int dB  = 128 + w * 16;                      // cols 128..255
  int cb0 = blockIdx.y * 4096;                 // split-K chunk of 262144 contraction
  v8f cA = {}, cB = {};
  for (int s = 0; s < 128; ++s) {
    int kb = cb0 + 32 * s;
    v16bf a  = load_a_frag(E,   nl0, CTOT, kb, lane);
    v16bf bA = load_b_frag(Vbt, dA,  CTOT, kb, lane);
    cA = wmma_bf16(a, bA, cA);
    v16bf bB = load_b_frag(Vbt, dB,  CTOT, kb, lane);
    cB = wmma_bf16(a, bB, cB);
  }
  int r = lane & 15, h = lane >> 4;
#pragma unroll
  for (int v = 0; v < 8; ++v) {
    size_t rowoff = (size_t)(n0 + nl0 + v + 8*h) * DEM;
    atomicAdd(&out[rowoff + dA + r], cA[v]);
    atomicAdd(&out[rowoff + dB + r], cB[v]);
  }
}

// ---------------- host launcher ----------------
extern "C" void kernel_launch(void* const* d_in, const int* in_sizes, int n_in,
                              void* d_out, int out_size, void* d_ws, size_t ws_size,
                              hipStream_t stream) {
  const float* X   = (const float*)d_in[0];
  const float* Wqw = (const float*)d_in[1];
  const float* Wqb = (const float*)d_in[2];
  const float* Wkw = (const float*)d_in[3];
  const float* Wkb = (const float*)d_in[4];
  const float* T   = (const float*)d_in[5];
  const float* V   = (const float*)d_in[6];
  float* out = (float*)d_out;
  (void)in_sizes; (void)n_in; (void)out_size; (void)ws_size;

  char* ws = (char*)d_ws;
  size_t off = 0;
  auto alloc = [&](size_t bytes) {
    void* p = ws + off;
    off = (off + bytes + 255) & ~(size_t)255;
    return p;
  };
  bf16_t* Qb   = (bf16_t*)alloc((size_t)SEQ * DEM * 2);   // 0.5 MB
  bf16_t* Kb   = (bf16_t*)alloc((size_t)SEQ * DEM * 2);   // 0.5 MB
  bf16_t* Tb2  = (bf16_t*)alloc((size_t)JK  * DEM * 2);   // 32 MB  [k][j][i]
  bf16_t* Vbt  = (bf16_t*)alloc((size_t)DEM * CTOT * 2);  // 128 MB [d][(m,k)]
  bf16_t* tmpt = (bf16_t*)alloc((size_t)NT  * JK  * 2);   // 8 MB   [nl][k][j]
  bf16_t* E    = (bf16_t*)alloc((size_t)NT  * CTOT * 2);  // 32 MB  [nl][m][k]
  float*  Z    = (float*) alloc((size_t)NT  * DEM * 4);   // 64 KB

  zero_f32<<<(SEQ * DEM + 255) / 256, 256, 0, stream>>>(out, SEQ * DEM);
  qk_kernel<<<SEQ, DEM, 0, stream>>>(X, Wqw, Wqb, Wkw, Wkb, Qb, Kb);
  transpose_T<<<dim3(JK / 32, DEM / 32), dim3(32, 8), 0, stream>>>(T, Tb2);
  transpose_V<<<dim3(CTOT / 32, DEM / 32), dim3(32, 8), 0, stream>>>(V, Vbt);

  for (int t = 0; t < SEQ / NT; ++t) {
    int n0 = t * NT;
    phase2_kernel<<<dim3(NT / 16, JK / 128), 256, 0, stream>>>(Qb, Tb2, tmpt, n0);
    zero_f32<<<(NT * DEM + 255) / 256, 256, 0, stream>>>(Z, NT * DEM);
    phase3a_kernel<<<dim3(SEQ / 64, DEM / 64, NT), 256, 0, stream>>>(Kb, tmpt, E, Z);
    normalize_kernel<<<(NT * (CTOT / 8)) / 256, 256, 0, stream>>>(E, Z);
    phase3b_kernel<<<dim3(NT / 16, 64), 256, 0, stream>>>(E, Vbt, out, n0);
  }
}
